// SAGE_88287347737171
// MI455X (gfx1250) — matrix operand order repaired and checked by reading, not compiled, not run on hardware
//
#include <hip/hip_runtime.h>

#define NNODES 50000
#define NEDGES 800000
#define DH 128
#define BN_EPS 1e-5f

typedef __attribute__((ext_vector_type(16))) __bf16 v16bf;
typedef __attribute__((ext_vector_type(8)))  float  v8f;

// ---- guaranteed hardware no-return f32 atomic add (L2 atomic unit) ----
__device__ __forceinline__ void gadd_f32(float* p, float v) {
    asm volatile("global_atomic_add_f32 %0, %1, off" :: "v"(p), "v"(v) : "memory");
}

// ---------------- utility kernels ----------------
__global__ void zero_f32_kernel(float* __restrict__ p, int n) {
    int i = blockIdx.x * blockDim.x + threadIdx.x;
    if (i < n) p[i] = 0.0f;
}

__global__ void cvt_bf16_kernel(const float* __restrict__ src, __bf16* __restrict__ dst, int n) {
    int i = blockIdx.x * blockDim.x + threadIdx.x;
    if (i < n) dst[i] = (__bf16)src[i];
}

__global__ void deg_count_kernel(const int* __restrict__ dst, float* __restrict__ deg, int nE) {
    int e = blockIdx.x * blockDim.x + threadIdx.x;
    if (e < nE) gadd_f32(&deg[dst[e]], 1.0f);
}

__global__ void deg_finalize_kernel(float* __restrict__ deg, int n) {
    int i = blockIdx.x * blockDim.x + threadIdx.x;
    if (i < n) deg[i] = 1.0f / fmaxf(deg[i], 1.0f);
}

// ---- scatter-add: agg[dst] += x[src], one thread = (edge, 4-float chunk) ----
__global__ void scatter_add_kernel(const float* __restrict__ x, const int* __restrict__ src,
                                   const int* __restrict__ dst, float* __restrict__ agg, int nE) {
    int tid = blockIdx.x * blockDim.x + threadIdx.x;
    int e = tid >> 5;              // 32 chunks of 4 floats cover DH=128
    if (e >= nE) return;
    int c = (tid & 31) << 2;
    int s = src[e], d = dst[e];
    const float4 v = *(const float4*)(x + (size_t)s * DH + c);
    float* p = agg + (size_t)d * DH + c;
    gadd_f32(p + 0, v.x);
    gadd_f32(p + 1, v.y);
    gadd_f32(p + 2, v.z);
    gadd_f32(p + 3, v.w);
}

// ---- fused SAGE linear: out = (agg*inv_deg) @ Wl^T + bl + xin @ Wr^T ----
// One wave per 16-row tile; K = 128 = 4 WMMA k-steps per operand.
// A layout (16-bit A 16x32): lane L, g=L>>4, row=L&15; elems 0..15 -> K=kb+16g .. kb+16g+15
// B layout (32x16): lane L = column L&15; elems 0..15 -> K = kb+16g .. kb+16g+15
//   (contiguous 32-byte read of a pre-converted bf16 W row)
// D layout (f32 16x16): acc[v] -> row v+8g, col L&15
template <int DOUT>
__global__ __launch_bounds__(256) void sage_gemm_kernel(
        const float* __restrict__ agg, const float* __restrict__ xin,
        const float* __restrict__ inv_deg,
        const __bf16* __restrict__ Wl, const float* __restrict__ bl,
        const __bf16* __restrict__ Wr, float* __restrict__ out) {
    constexpr int K = DH;
    constexpr int NT = DOUT / 16;
    int wave = blockIdx.x * (blockDim.x >> 5) + (threadIdx.x >> 5);
    if (wave >= NNODES / 16) return;          // whole wave exits: EXEC stays all-1s for WMMA
    int lane = threadIdx.x & 31;
    int rlo  = lane & 15;                      // A row-in-tile; also B/D column-in-tile
    int g    = lane >> 4;
    int row  = wave * 16 + rlo;

    const float s = inv_deg[row];
    const float* ag = agg + (size_t)row * K;
    const float* xg = xin + (size_t)row * K;

    v16bf Am[4], Ax[4];
#pragma unroll
    for (int kk = 0; kk < 4; ++kk) {
        int kb = kk * 32 + g * 16;
        const float* p0 = ag + kb;
        const float* p1 = ag + kb + 8;
        const float* q0 = xg + kb;
        const float* q1 = xg + kb + 8;
#pragma unroll
        for (int i = 0; i < 8; ++i) {
            Am[kk][i]     = (__bf16)(p0[i] * s);
            Am[kk][8 + i] = (__bf16)(p1[i] * s);
            Ax[kk][i]     = (__bf16)q0[i];
            Ax[kk][8 + i] = (__bf16)q1[i];
        }
    }

#pragma unroll
    for (int n = 0; n < NT; ++n) {
        int wrow = n * 16 + rlo;               // output column == W row (out = in @ W^T)
        const __bf16* wl = Wl + (size_t)wrow * K;
        const __bf16* wr = Wr + (size_t)wrow * K;
        v8f acc = {0.f, 0.f, 0.f, 0.f, 0.f, 0.f, 0.f, 0.f};
#pragma unroll
        for (int kk = 0; kk < 4; ++kk) {
            int kb = kk * 32 + g * 16;
            v16bf B = *(const v16bf*)(wl + kb);   // 32B aligned: kb multiple of 16 elems
            acc = __builtin_amdgcn_wmma_f32_16x16x32_bf16(false, Am[kk], false, B,
                                                          (short)0, acc, false, false);
        }
#pragma unroll
        for (int kk = 0; kk < 4; ++kk) {
            int kb = kk * 32 + g * 16;
            v16bf B = *(const v16bf*)(wr + kb);
            acc = __builtin_amdgcn_wmma_f32_16x16x32_bf16(false, Ax[kk], false, B,
                                                          (short)0, acc, false, false);
        }
        float bias = bl[n * 16 + rlo];
#pragma unroll
        for (int v = 0; v < 8; ++v) {
            int m = v + 8 * g;
            out[(size_t)(wave * 16 + m) * DOUT + n * 16 + rlo] = acc[v] + bias;
        }
    }
}

// ---- BN column stats: stats[0..127]=sum, stats[128..255]=sumsq ----
__global__ void bn_stats_kernel(const float* __restrict__ h, float* __restrict__ stats, int n) {
    int tid = blockIdx.x * blockDim.x + threadIdx.x;
    int col = tid & (DH - 1);
    int r0  = tid >> 7;
    int rs  = (gridDim.x * blockDim.x) >> 7;
    float s = 0.f, s2 = 0.f;
    for (int r = r0; r < n; r += rs) {
        float v = h[(size_t)r * DH + col];
        s += v; s2 += v * v;
    }
    gadd_f32(&stats[col], s);
    gadd_f32(&stats[DH + col], s2);
}

__global__ void bn_apply_relu_kernel(const float* __restrict__ in, const float* __restrict__ stats,
                                     const float* __restrict__ gamma, const float* __restrict__ beta,
                                     float* __restrict__ out, int n) {
    int idx = blockIdx.x * blockDim.x + threadIdx.x;
    if (idx >= n * DH) return;
    int col = idx & (DH - 1);
    const float invN = 1.0f / (float)NNODES;
    float mu  = stats[col] * invN;
    float var = stats[DH + col] * invN - mu * mu;
    float sc  = gamma[col] * rsqrtf(var + BN_EPS);
    float v   = (in[idx] - mu) * sc + beta[col];
    out[idx]  = v > 0.f ? v : 0.f;
}

// ---------------- launch ----------------
extern "C" void kernel_launch(void* const* d_in, const int* in_sizes, int n_in,
                              void* d_out, int out_size, void* d_ws, size_t ws_size,
                              hipStream_t stream) {
    const float* x   = (const float*)d_in[0];
    const int*   ei  = (const int*)d_in[1];
    const int*   src = ei;
    const int*   dst = ei + NEDGES;
    const float *Wl0 = (const float*)d_in[2],  *bl0 = (const float*)d_in[3],
                *Wr0 = (const float*)d_in[4],  *g0  = (const float*)d_in[5],
                *be0 = (const float*)d_in[6],
                *Wl1 = (const float*)d_in[7],  *bl1 = (const float*)d_in[8],
                *Wr1 = (const float*)d_in[9],  *g1  = (const float*)d_in[10],
                *be1 = (const float*)d_in[11],
                *Wl2 = (const float*)d_in[12], *bl2 = (const float*)d_in[13],
                *Wr2 = (const float*)d_in[14];
    float* fout = (float*)d_out;

    // ws layout (all chunks 32B aligned)
    float*  bufA  = (float*)d_ws;                      // agg / pre-BN / h1
    float*  bufB  = bufA + (size_t)NNODES * DH;        // h0 / layer2 agg
    __bf16* wbl0  = (__bf16*)(bufB + (size_t)NNODES * DH);
    __bf16* wbr0  = wbl0 + DH * DH;
    __bf16* wbl1  = wbr0 + DH * DH;
    __bf16* wbr1  = wbl1 + DH * DH;
    __bf16* wbl2  = wbr1 + DH * DH;
    __bf16* wbr2  = wbl2 + 64 * DH;
    float*  ideg  = (float*)(wbr2 + 64 * DH);
    float*  stats = ideg + NNODES;

    const int TPB = 256;
    const int featBlocks    = (NNODES * DH + TPB - 1) / TPB;       // 25000
    const int nodeBlocks    = (NNODES + TPB - 1) / TPB;
    const int edgeBlocks    = (NEDGES + TPB - 1) / TPB;
    const int scatterBlocks = (NEDGES * 32 + TPB - 1) / TPB;       // 100000
    const int gemmBlocks    = (NNODES / 16 + 7) / 8;               // 8 waves/block
    const int wBlocks       = (DH * DH + TPB - 1) / TPB;           // 64
    const int wBlocks2      = (64 * DH + TPB - 1) / TPB;           // 32

    // one-time (per launch) bf16 weight conversion — tiny, L2 resident
    cvt_bf16_kernel<<<wBlocks,  TPB, 0, stream>>>(Wl0, wbl0, DH * DH);
    cvt_bf16_kernel<<<wBlocks,  TPB, 0, stream>>>(Wr0, wbr0, DH * DH);
    cvt_bf16_kernel<<<wBlocks,  TPB, 0, stream>>>(Wl1, wbl1, DH * DH);
    cvt_bf16_kernel<<<wBlocks,  TPB, 0, stream>>>(Wr1, wbr1, DH * DH);
    cvt_bf16_kernel<<<wBlocks2, TPB, 0, stream>>>(Wl2, wbl2, 64 * DH);
    cvt_bf16_kernel<<<wBlocks2, TPB, 0, stream>>>(Wr2, wbr2, 64 * DH);

    // degree -> 1/max(deg,1)
    zero_f32_kernel<<<nodeBlocks, TPB, 0, stream>>>(ideg, NNODES);
    deg_count_kernel<<<edgeBlocks, TPB, 0, stream>>>(dst, ideg, NEDGES);
    deg_finalize_kernel<<<nodeBlocks, TPB, 0, stream>>>(ideg, NNODES);

    // ---- layer 0 ----
    zero_f32_kernel<<<featBlocks, TPB, 0, stream>>>(bufA, NNODES * DH);
    scatter_add_kernel<<<scatterBlocks, TPB, 0, stream>>>(x, src, dst, bufA, NEDGES);
    sage_gemm_kernel<128><<<gemmBlocks, TPB, 0, stream>>>(bufA, x, ideg, wbl0, bl0, wbr0, bufA);
    zero_f32_kernel<<<1, TPB, 0, stream>>>(stats, 2 * DH);
    bn_stats_kernel<<<512, TPB, 0, stream>>>(bufA, stats, NNODES);
    bn_apply_relu_kernel<<<featBlocks, TPB, 0, stream>>>(bufA, stats, g0, be0, bufB, NNODES);

    // ---- layer 1 ----
    zero_f32_kernel<<<featBlocks, TPB, 0, stream>>>(bufA, NNODES * DH);
    scatter_add_kernel<<<scatterBlocks, TPB, 0, stream>>>(bufB, src, dst, bufA, NEDGES);
    sage_gemm_kernel<128><<<gemmBlocks, TPB, 0, stream>>>(bufA, bufB, ideg, wbl1, bl1, wbr1, bufA);
    zero_f32_kernel<<<1, TPB, 0, stream>>>(stats, 2 * DH);
    bn_stats_kernel<<<512, TPB, 0, stream>>>(bufA, stats, NNODES);
    bn_apply_relu_kernel<<<featBlocks, TPB, 0, stream>>>(bufA, stats, g1, be1, bufA, NNODES);

    // ---- layer 2 (D_OUT = 64, straight to d_out) ----
    zero_f32_kernel<<<featBlocks, TPB, 0, stream>>>(bufB, NNODES * DH);
    scatter_add_kernel<<<scatterBlocks, TPB, 0, stream>>>(bufA, src, dst, bufB, NEDGES);
    sage_gemm_kernel<64><<<gemmBlocks, TPB, 0, stream>>>(bufB, bufA, ideg, wbl2, bl2, wbr2, fout);
}
